// QuantumKernelAttention_65481071408018
// MI455X (gfx1250) — matrix operand order repaired and checked by reading, not compiled, not run on hardware
//
#include <hip/hip_runtime.h>

// QuantumKernelAttention for MI455X (gfx1250, wave32, WMMA).
// B=8, S=1024, E=512, H=8, dk=64, 8 "wires".
//
// Key transforms vs reference:
//  1. cos(0.5(a-b)) = cos(.5a)cos(.5b) + sin(.5a)sin(.5b)  -> precompute a
//     (B*H, S, 16) f32 cos/sin table once (1M sincos instead of 537M cos).
//  2. Row max of |prod cos| is exactly 1 (diagonal), so softmax is a single
//     pass: P = exp(score-1), denom = row-sum. Flash-style, scores never hit
//     global memory.
//  3. P@V (8.6 GFLOP) runs on v_wmma_f32_16x16x32_f16.
//  4. V tile staged into LDS *transposed* (dim-major) so each B operand
//     half loads as one contiguous ds_load_b128 (no per-element gather).
//  5. Score math computed on <2 x float> pairs (K=j and K=j+16) so the
//     backend can select v_pk_fma_f32 / v_pk_mul_f32.
//  6. kcs tile copied with global_load_async_to_lds_b64 (ASYNCcnt) when the
//     toolchain exposes the builtin; plain load+store fallback otherwise.

#define B_  8
#define S_  1024
#define E_  512
#define H_  8
#define DK  64
#define NW  8
#define VPAD 40   // padded key-dim of vt2: 80B row stride = 20 words
                  // -> 20*n mod 64 injective for n=0..15 (conflict-free),
                  // and 80 = 5*16 keeps 16B alignment for b128 loads.

#if defined(__has_builtin)
#if __has_builtin(__builtin_amdgcn_global_load_async_to_lds_b64)
#define USE_ASYNC_LDS 1
#endif
#endif

typedef __attribute__((ext_vector_type(16))) _Float16 v16h;
typedef __attribute__((ext_vector_type(8)))  _Float16 v8h;
typedef __attribute__((ext_vector_type(4)))  _Float16 v4h;
typedef __attribute__((ext_vector_type(8)))  float    v8f;
typedef __attribute__((ext_vector_type(2)))  float    v2f;

// Exact pointee type the async builtin wants: <2 x i32> (gcc vector_size).
typedef int b64vec __attribute__((vector_size(8)));
typedef __attribute__((address_space(1))) b64vec* gptr_b64;
typedef __attribute__((address_space(3))) b64vec* lptr_b64;

// One packed "wire" term for two key columns (j, j+16):
// t = qc*kc + qs*ks  (candidate for v_pk_fma_f32 + v_pk_mul_f32)
__device__ __forceinline__ v2f term2(float qcw, float qsw,
                                     float kca, float kcb,
                                     float ksa, float ksb) {
  v2f kc = {kca, kcb}, ks = {ksa, ksb};
  v2f qcv = {qcw, qcw}, qsv = {qsw, qsw};
  return __builtin_elementwise_fma(qcv, kc, qsv * ks);
}

// ---------------- Kernel 1: cos/sin table ----------------
// cs[(b*H+h)*S + s][0..7] = cos(0.5*x_w), [8..15] = sin(0.5*x_w)
__global__ __launch_bounds__(256) void qk_cs_precompute(
    const float* __restrict__ x, float* __restrict__ cs)
{
  int idx = blockIdx.x * blockDim.x + threadIdx.x;   // ((b*H+h)*S+s)*8+w
  if (idx >= B_ * H_ * S_ * NW) return;
  int w = idx & 7;
  int s = (idx >> 3) & (S_ - 1);
  int h = (idx >> 13) & 7;
  int b = idx >> 16;
  float v = 0.5f * x[((size_t)b * S_ + s) * E_ + h * DK + w];
  float* dst = cs + (((size_t)(b * H_ + h)) * S_ + s) * 16;
  dst[w]     = __cosf(v);
  dst[8 + w] = __sinf(v);
}

// ---------------- Kernel 2: flash attention with WMMA ----------------
// grid: (B*H, S/128). block: 256 (8 waves). Wave w owns rows
// [blockIdx.y*128 + w*16, +16). All waves share the per-key-block LDS tiles.
__global__ __launch_bounds__(256) void qk_attn(
    const float* __restrict__ x, const float* __restrict__ cs,
    float* __restrict__ out)
{
  __shared__ float    kcs[32][16];        // cos/sin of 32 keys (2 KB)
  __shared__ _Float16 vt2[DK][VPAD];      // V tile TRANSPOSED: [dim][key] (5 KB)

  const int bh   = blockIdx.x;            // 0..63
  const int b    = bh >> 3;
  const int h    = bh & 7;
  const int lane = threadIdx.x & 31;
  const int half = lane >> 4;             // 0: lanes 0-15, 1: lanes 16-31
  const int l15  = lane & 15;
  const int row0 = blockIdx.y * 128 + (threadIdx.x >> 5) * 16;
  const int myrow = row0 + l15;           // this lane's query row (both halves)

  // Staging coordinates.
  const int d2 = (threadIdx.x & 31) * 2;  // dims {d2, d2+1}
  const int jg = threadIdx.x >> 5;        // keys {4jg .. 4jg+3}
  const int cj = threadIdx.x >> 3;        // kcs: key row for this thread
  const int cp = (threadIdx.x & 7) * 2;   // kcs: float pair within row

  // Query cos/sin: 16 contiguous f32 for row `myrow`.
  float qc[8], qs[8];
  {
    const float4* qp = (const float4*)(cs + (((size_t)bh) * S_ + myrow) * 16);
    float4 a0 = qp[0], a1 = qp[1], c0 = qp[2], c1 = qp[3];
    qc[0]=a0.x; qc[1]=a0.y; qc[2]=a0.z; qc[3]=a0.w;
    qc[4]=a1.x; qc[5]=a1.y; qc[6]=a1.z; qc[7]=a1.w;
    qs[0]=c0.x; qs[1]=c0.y; qs[2]=c0.z; qs[3]=c0.w;
    qs[4]=c1.x; qs[5]=c1.y; qs[6]=c1.z; qs[7]=c1.w;
  }

  v8f acc[4] = {};                        // 16x64 f32 accumulator (4 WMMA tiles)
  float psum = 0.0f;                      // partial row denominator

  for (int kb = 0; kb < S_; kb += 32) {
    __syncthreads();                      // protect LDS from previous iteration

    // --- Stage K cos/sin tile: branchless, 8 bytes per thread ---
    {
      const float* src = cs + (((size_t)bh) * S_ + kb + cj) * 16 + cp;
#ifdef USE_ASYNC_LDS
      __builtin_amdgcn_global_load_async_to_lds_b64(
          (gptr_b64)(float*)src, (lptr_b64)&kcs[cj][cp], 0, 0);
#else
      *(float2*)&kcs[cj][cp] = *(const float2*)src;
#endif
    }
    // --- Stage V tile transposed: 4x global_load_b64 + 2x ds_store_b64 ---
    {
      float2 c[4];
      #pragma unroll
      for (int q = 0; q < 4; ++q)
        c[q] = *(const float2*)(x + ((size_t)b * S_ + kb + jg * 4 + q) * E_ +
                                h * DK + d2);
      v4h r0, r1;
      #pragma unroll
      for (int q = 0; q < 4; ++q) {
        r0[q] = (_Float16)c[q].x;
        r1[q] = (_Float16)c[q].y;
      }
      *(v4h*)&vt2[d2][jg * 4]     = r0;
      *(v4h*)&vt2[d2 + 1][jg * 4] = r1;
      // Prefetch next key block's V rows into cache (global_prefetch_b8).
      if (kb + 32 < S_)
        __builtin_prefetch(x + ((size_t)b * S_ + kb + 32 + jg * 4) * E_ +
                           h * DK + d2, 0, 1);
    }
#ifdef USE_ASYNC_LDS
#if defined(__has_builtin) && __has_builtin(__builtin_amdgcn_s_wait_asynccnt)
    __builtin_amdgcn_s_wait_asynccnt(0);
#else
    asm volatile("s_wait_asynccnt 0" ::: "memory");
#endif
#endif
    __syncthreads();

    // --- Build P tile (16x32 f16) directly in WMMA A-operand layout.
    // ISA 7.12.2 (16-bit A 16x32): lanes 0-15 row M=l15, K in {0..7,16..23};
    // lanes 16-31 row M=l15, K in {8..15,24..31}. v16h element e -> VGPR e/2
    // (lo/hi): K(e) = (e&7) + 8*half + 16*(e>>3). Elements e and e+8 share q
    // values -> compute as <2 x float> pairs (keys j0 and j0+16).
    v16h a;
    #pragma unroll
    for (int e = 0; e < 8; ++e) {
      const int j0 = e + (half << 3);
      const float4* ka = (const float4*)&kcs[j0][0];
      const float4* kz = (const float4*)&kcs[j0 + 16][0];
      float4 ac0 = ka[0], ac1 = ka[1], as0 = ka[2], as1 = ka[3];
      float4 zc0 = kz[0], zc1 = kz[1], zs0 = kz[2], zs1 = kz[3];
      v2f t0 = term2(qc[0], qs[0], ac0.x, zc0.x, as0.x, zs0.x);
      v2f t1 = term2(qc[1], qs[1], ac0.y, zc0.y, as0.y, zs0.y);
      v2f t2 = term2(qc[2], qs[2], ac0.z, zc0.z, as0.z, zs0.z);
      v2f t3 = term2(qc[3], qs[3], ac0.w, zc0.w, as0.w, zs0.w);
      v2f t4 = term2(qc[4], qs[4], ac1.x, zc1.x, as1.x, zs1.x);
      v2f t5 = term2(qc[5], qs[5], ac1.y, zc1.y, as1.y, zs1.y);
      v2f t6 = term2(qc[6], qs[6], ac1.z, zc1.z, as1.z, zs1.z);
      v2f t7 = term2(qc[7], qs[7], ac1.w, zc1.w, as1.w, zs1.w);
      v2f prod = ((t0 * t1) * (t2 * t3)) * ((t4 * t5) * (t6 * t7));
      float p0 = __expf(__builtin_fabsf(prod.x) - 1.0f);  // softmax max == 1
      float p1 = __expf(__builtin_fabsf(prod.y) - 1.0f);
      psum += p0 + p1;
      a[e]     = (_Float16)p0;   // K = j0
      a[e + 8] = (_Float16)p1;   // K = j0 + 16
    }

    // --- B tiles (32x16 f16) via contiguous b128 loads from transposed LDS.
    // B layout: lane col N = l15; element e -> K = (e&7) + 8*half + 16*(e>>3)
    // (matches A). With vt2[dim][key], lane's elements are
    // vt2[t*16+l15][8*half .. +7] and [16+8*half .. +7] -> two ds_load_b128.
    #pragma unroll
    for (int t = 0; t < 4; ++t) {
      const v8h* p = (const v8h*)&vt2[t * 16 + l15][8 * half];
      v8h lo = p[0], hi = p[2];
      v16h bt;
      #pragma unroll
      for (int e = 0; e < 8; ++e) {
        bt[e]     = lo[e];
        bt[8 + e] = hi[e];
      }
      acc[t] = __builtin_amdgcn_wmma_f32_16x16x32_f16(
          false, a, false, bt, (short)0, acc[t], false, false);
    }
  }

  // Row denominators: row m is split across lanes m and m+16.
  psum += __shfl_xor(psum, 16, 32);          // wave32 pair reduce
  // C/D layout: VGPR r of lane group `half` holds row M = r + 8*half.
  float rden[8];
  #pragma unroll
  for (int r = 0; r < 8; ++r)
    rden[r] = 1.0f / __shfl(psum, r + 8 * half, 32);

  // Store: out[b, row, h*64 + t*16 + l15]
  #pragma unroll
  for (int t = 0; t < 4; ++t) {
    #pragma unroll
    for (int r = 0; r < 8; ++r) {
      int srow = row0 + r + 8 * half;
      out[((size_t)b * S_ + srow) * E_ + h * DK + t * 16 + l15] =
          acc[t][r] * rden[r];
    }
  }
}

extern "C" void kernel_launch(void* const* d_in, const int* in_sizes, int n_in,
                              void* d_out, int out_size, void* d_ws, size_t ws_size,
                              hipStream_t stream) {
  const float* x  = (const float*)d_in[0];
  float* out      = (float*)d_out;
  float* cs       = (float*)d_ws;   // needs 64*1024*16*4 = 4 MB

  qk_cs_precompute<<<dim3((B_ * H_ * S_ * NW) / 256), 256, 0, stream>>>(x, cs);
  qk_attn<<<dim3(B_ * H_, S_ / 128), 256, 0, stream>>>(x, cs, out);
}
